// SE3TBackbone_74019466379360
// MI455X (gfx1250) — matrix operand order repaired and checked by reading, not compiled, not run on hardware
//
#include <hip/hip_runtime.h>
#include <hip/hip_bf16.h>

#define NB 64
#define NA 64
#define NN 4096        // NB*NA nodes
#define EPB 4032       // NA*(NA-1) edges per molecule
#define NE 258048      // NB*EPB total edges
#define NTILE 16128    // NE/16 edge tiles
#define BCHUNK 2112    // uint4s per staged B chain: 33 m-slots * 32 lanes * 32B / 16

typedef __attribute__((ext_vector_type(16))) __bf16 v16bf;
typedef __attribute__((ext_vector_type(8)))  float  v8f;

// ---------------- device helpers ----------------

__device__ __forceinline__ float wsum32(float v) {
#pragma unroll
  for (int m = 16; m; m >>= 1) v += __shfl_xor(v, m, 32);
  return v;
}

__device__ __forceinline__ void edge_decode(int e, int& srcn, int& dstn) {
  int b = e / EPB, rem = e - b * EPB;
  int s = rem / 63, dd = rem - s * 63;
  int d = dd + (dd >= s ? 1 : 0);
  srcn = b * NA + s;
  dstn = b * NA + d;
}

// A-fragment (bf16 16x32): lane l (row=l&15, khalf=l>>4) holds K = {8k..8k+7, 16+8k..+7}
__device__ __forceinline__ v16bf load_ffragb(const __bf16* F, int e, int kh) {
  const uint4* p0 = (const uint4*)(F + e * 32 + kh * 8);
  const uint4* p1 = (const uint4*)(F + e * 32 + kh * 8 + 16);
  union { uint4 u[2]; v16bf v; } t;
  t.u[0] = *p0; t.u[1] = *p1;
  return t.v;
}

__device__ __forceinline__ v16bf scale_frag(v16bf f, __bf16 s) {
  v16bf a;
#pragma unroll
  for (int j = 0; j < 16; ++j) a[j] = f[j] * s;
  return a;
}

// cooperative stage of one radial's B chain (33KB, contiguous per colblock) into LDS
__device__ __forceinline__ void stageB(const __bf16* Bp, int cb, uint4* bsh, int tid) {
  const uint4* g = (const uint4*)Bp + (size_t)cb * BCHUNK;
  for (int i = tid; i < BCHUNK; i += 128) bsh[i] = g[i];
}

__device__ __forceinline__ void prefetchB(const __bf16* Bp, int cb, int tid) {
  const char* p = (const char*)((const uint4*)Bp + (size_t)cb * BCHUNK);
  int off = tid * 256;
  if (off < BCHUNK * 16) {
    __builtin_prefetch(p + off, 0, 1);        // -> global_prefetch_b8
    __builtin_prefetch(p + off + 128, 0, 1);
  }
}

// acc += Sum_m WMMA(hh[:,m] * feat, B_m) ; m==32 slot is the bias row (hh=1). B from LDS.
__device__ __forceinline__ v8f chainL(const __bf16* hh, const uint4* bsh, v16bf frag,
                                      v8f acc, int lane, int eA) {
  for (int m = 0; m <= 32; ++m) {
    __bf16 s = (m < 32) ? hh[eA * 32 + m] : (__bf16)1.0f;
    v16bf a = scale_frag(frag, s);
    const v16bf* bp = (const v16bf*)(bsh + m * 64 + lane * 2);
    acc = __builtin_amdgcn_wmma_f32_16x16x32_bf16(false, a, false, *bp, (short)0, acc,
                                                  false, false);
  }
  return acc;
}

// ---------------- kernels ----------------

__global__ __launch_bounds__(256) void geom_kernel(const float* x, float* r, float* dirv) {
  int e = blockIdx.x * 256 + threadIdx.x;
  if (e >= NE) return;
  int sn, dn; edge_decode(e, sn, dn);
  const float* xs = x + sn * 3;
  const float* xd = x + dn * 3;
  float rx = xs[0] - xd[0], ry = xs[1] - xd[1], rz = xs[2] - xd[2];
  float rr = sqrtf(rx * rx + ry * ry + rz * rz + 1e-12f);
  r[e] = rr;
  float inv = 1.0f / rr;
  dirv[e * 3 + 0] = rx * inv; dirv[e * 3 + 1] = ry * inv; dirv[e * 3 + 2] = rz * inv;
}

// radial hidden: hh[e][m] = relu(LN(r*w1+b1)); one wave (32 lanes = MID) per edge
__global__ __launch_bounds__(256) void hh_kernel(const float* r, const float* w1, const float* b1,
                                                 const float* g, const float* be, __bf16* dst) {
  int gt = blockIdx.x * 256 + threadIdx.x;
  int e = gt >> 5, m = gt & 31;
  if (e >= NE) return;
  float t = r[e] * w1[m] + b1[m];
  float mu = wsum32(t) * (1.0f / 32.0f);
  float dv = t - mu;
  float var = wsum32(dv * dv) * (1.0f / 32.0f);
  float h = dv * rsqrtf(var + 1e-5f) * g[m] + be[m];
  dst[e * 32 + m] = (__bf16)fmaxf(h, 0.0f);
}

// pack w2 (+bias slot m==32) into WMMA B-fragment layout, contiguous per colblock,
// K zero-padded to 32.  dst elem index = ((cb*33 + m)*32 + lane)*16
__global__ __launch_bounds__(32) void pack_b_kernel(const float* w2, const float* b2, __bf16* dst,
                                                    int cin, int cout, int ncb) {
  int m = blockIdx.x / ncb, cb = blockIdx.x % ncb;
  int lane = threadIdx.x;
  int col = cb * 16 + (lane & 15);
  int k0 = (lane >> 4) * 8;
  __bf16* o = dst + ((((cb * 33 + m) * 32) + lane) << 4);
#pragma unroll
  for (int j = 0; j < 16; ++j) {
    int i = (j < 8) ? k0 + j : 16 + k0 + (j - 8);
    float v = 0.0f;
    if (i < cin && col < cout) v = (m < 32) ? w2[(m * cout + col) * cin + i] : b2[col * cin + i];
    o[j] = (__bf16)v;
  }
}

// gather per-edge features into bf16 A-operand arrays (zero-padded to cin 32)
__global__ __launch_bounds__(256) void edge_feat_kernel(const float* f0, const float* f1,
                                                        const float* dirv,
                                                        __bf16* Ff0, __bf16* Fpr,
                                                        __bf16* F1x, __bf16* F1y, __bf16* F1z,
                                                        __bf16* Fcx, __bf16* Fcy, __bf16* Fcz,
                                                        int c0i, int has_f1) {
  int e = blockIdx.x * 256 + threadIdx.x;
  if (e >= NE) return;
  int sn, dn; edge_decode(e, sn, dn);
  float dx = dirv[e * 3], dy = dirv[e * 3 + 1], dz = dirv[e * 3 + 2];
  for (int i = 0; i < 32; ++i) {
    float v = (i < c0i) ? f0[sn * c0i + i] : 0.0f;
    Ff0[e * 32 + i] = (__bf16)v;
  }
  if (has_f1) {
    for (int i = 0; i < 32; ++i) {
      const float* fp = f1 + (sn * 32 + i) * 3;
      float fx = fp[0], fy = fp[1], fz = fp[2];
      F1x[e * 32 + i] = (__bf16)fx; F1y[e * 32 + i] = (__bf16)fy; F1z[e * 32 + i] = (__bf16)fz;
      Fpr[e * 32 + i] = (__bf16)(fx * dx + fy * dy + fz * dz);
      Fcx[e * 32 + i] = (__bf16)(dy * fz - dz * fy);
      Fcy[e * 32 + i] = (__bf16)(dz * fx - dx * fz);
      Fcz[e * 32 + i] = (__bf16)(dx * fy - dy * fx);
    }
  }
}

// the heavy kernel: 4 waves per block, each on its own 16-edge tile, sharing one
// LDS-staged B chain per radial (33KB); WMMA chains build m0/m1
struct ConvArgs {
  const __bf16 *hh00, *hh01, *hh10, *hh11a, *hh11b;
  const __bf16 *B00, *B01, *B10, *B11a, *B11b;
  const __bf16 *Ff0, *Fpr, *F1x, *F1y, *F1z, *Fcx, *Fcy, *Fcz;
  const float* dirv;
  float* out0;   // [E][cout]
  float* out1;   // [E][cout][3]
  int cout, ncb, has_f1, only0;
};

__global__ __launch_bounds__(128) void conv_kernel(ConvArgs A) {
  __shared__ uint4 bsh[BCHUNK];    // 33KB staged B chain
  int ncb = A.ncb;
  int grp = blockIdx.x / ncb, cb = blockIdx.x % ncb;
  int tid = threadIdx.x;
  int wid = tid >> 5, lane = tid & 31;
  int tile = grp * 4 + wid;
  int rA = lane & 15, kh = lane >> 4;
  int eA = tile * 16 + rA;
  int col = cb * 16 + rA;
  const v8f vzero = {0, 0, 0, 0, 0, 0, 0, 0};

  v16bf f0f = load_ffragb(A.Ff0, eA, kh);

  // ---- degree-0 output: r00(f0s) [+ r10(proj)] ----
  stageB(A.B00, cb, bsh, tid);
  if (!A.only0) prefetchB(A.B01, cb, tid);
  __syncthreads();
  v8f acc = chainL(A.hh00, bsh, f0f, vzero, lane, eA);
  v16bf prf;
  if (A.has_f1) {
    prf = load_ffragb(A.Fpr, eA, kh);
    __syncthreads();
    stageB(A.B10, cb, bsh, tid);
    __syncthreads();
    acc = chainL(A.hh10, bsh, prf, acc, lane, eA);
  }
#pragma unroll
  for (int v = 0; v < 8; ++v) {
    int eD = tile * 16 + v + kh * 8;
    A.out0[eD * A.cout + col] = acc[v];
  }
  if (A.only0) return;

  // ---- degree-1: s01 = r01(f0s); m1_c = s01*dir_c [+ r11a(f1_c) + r11b(cross_c)] ----
  __syncthreads();
  stageB(A.B01, cb, bsh, tid);
  if (A.has_f1) prefetchB(A.B11a, cb, tid);
  __syncthreads();
  v8f s01 = chainL(A.hh01, bsh, f0f, vzero, lane, eA);

  v8f m1[3] = {vzero, vzero, vzero};
  if (A.has_f1) {
    const __bf16* F1[3] = {A.F1x, A.F1y, A.F1z};
    const __bf16* FC[3] = {A.Fcx, A.Fcy, A.Fcz};
    __syncthreads();
    stageB(A.B11a, cb, bsh, tid);
    prefetchB(A.B11b, cb, tid);
    __syncthreads();
    for (int c = 0; c < 3; ++c)
      m1[c] = chainL(A.hh11a, bsh, load_ffragb(F1[c], eA, kh), m1[c], lane, eA);
    __syncthreads();
    stageB(A.B11b, cb, bsh, tid);
    __syncthreads();
    for (int c = 0; c < 3; ++c)
      m1[c] = chainL(A.hh11b, bsh, load_ffragb(FC[c], eA, kh), m1[c], lane, eA);
  }
  for (int c = 0; c < 3; ++c) {
#pragma unroll
    for (int v = 0; v < 8; ++v) {
      int eD = tile * 16 + v + kh * 8;
      A.out1[(eD * A.cout + col) * 3 + c] = m1[c][v] + s01[v] * A.dirv[eD * 3 + c];
    }
  }
}

// queries: q0 = f0 @ wq0 ; q1 = f1 contracted with wq1
__global__ __launch_bounds__(256) void q_kernel(const float* f0, const float* f1,
                                                const float* wq0, const float* wq1,
                                                float* q0, float* q1, int c0i, int has_f1) {
  int idx = blockIdx.x * 256 + threadIdx.x;
  if (idx >= NN * 32) return;
  int n = idx >> 5, o = idx & 31;
  float a = 0.0f;
  for (int i = 0; i < c0i; ++i) a += f0[n * c0i + i] * wq0[i * 32 + o];
  q0[idx] = a;
  if (has_f1) {
    float ax = 0, ay = 0, az = 0;
    for (int i = 0; i < 32; ++i) {
      float w = wq1[i * 32 + o];
      const float* fp = f1 + (n * 32 + i) * 3;
      ax += fp[0] * w; ay += fp[1] * w; az += fp[2] * w;
    }
    q1[idx * 3] = ax; q1[idx * 3 + 1] = ay; q1[idx * 3 + 2] = az;
  }
}

__global__ __launch_bounds__(256) void logits_kernel(const float* q0, const float* q1,
                                                     const float* k0, const float* k1,
                                                     float* logits, int has_f1, float scale) {
  int e = blockIdx.x * 256 + threadIdx.x;
  if (e >= NE) return;
  int sn, dn; edge_decode(e, sn, dn);
  for (int h = 0; h < 8; ++h) {
    float lg = 0.0f;
    for (int j = 0; j < 4; ++j) {
      int o = h * 4 + j;
      lg += q0[dn * 32 + o] * k0[e * 32 + o];
      if (has_f1) {
        const float* qp = q1 + (dn * 32 + o) * 3;
        const float* kp = k1 + (e * 32 + o) * 3;
        lg += qp[0] * kp[0] + qp[1] * kp[1] + qp[2] * kp[2];
      }
    }
    logits[e * 8 + h] = lg * scale;
  }
}

// per-destination segment softmax over 63 incoming edges
__global__ __launch_bounds__(64) void softmax_kernel(const float* logits, float* att) {
  __shared__ float buf[64];
  int n = blockIdx.x, b = n >> 6, d = n & 63;
  int t = threadIdx.x;
  int e = 0;
  if (t < 63) {
    int s = t + (t >= d ? 1 : 0);
    int dd = d - (d > s ? 1 : 0);
    e = b * EPB + s * 63 + dd;
  }
  for (int h = 0; h < 8; ++h) {
    float lg = (t < 63) ? logits[e * 8 + h] : -1e30f;
    buf[t] = lg; __syncthreads();
    for (int off = 32; off; off >>= 1) { if (t < off) buf[t] = fmaxf(buf[t], buf[t + off]); __syncthreads(); }
    float mx = buf[0]; __syncthreads();
    float ex = (t < 63) ? __expf(lg - mx) : 0.0f;
    buf[t] = ex; __syncthreads();
    for (int off = 32; off; off >>= 1) { if (t < off) buf[t] += buf[t + off]; __syncthreads(); }
    float sm = buf[0]; __syncthreads();
    if (t < 63) att[e * 8 + h] = ex / sm;
  }
}

// attention aggregation + skip-concat projection + SE(3) norm; one wave (lane=channel) per node
struct AggArgs {
  const float *att, *v0, *v1, *f0, *f1, *wo0, *wo1, *g0, *b0, *g1, *b1;
  float *f0o, *f1o;
  int c0i, has_f1;
};

__global__ __launch_bounds__(32) void aggregate_kernel(AggArgs A) {
  int n = blockIdx.x, b = n >> 6, d = n & 63;
  int o = threadIdx.x;
  __shared__ float sz0[16];
  __shared__ float sz1[16][3];
  if (o < 16) {
    int h = o >> 1;
    float z0 = 0, zx = 0, zy = 0, zz = 0;
    for (int sp = 0; sp < 63; ++sp) {
      int s = sp + (sp >= d ? 1 : 0);
      int dd = d - (d > s ? 1 : 0);
      int e = b * EPB + s * 63 + dd;
      float at = A.att[e * 8 + h];
      z0 += at * A.v0[e * 16 + o];
      const float* vp = A.v1 + (e * 16 + o) * 3;
      zx += at * vp[0]; zy += at * vp[1]; zz += at * vp[2];
    }
    sz0[o] = z0; sz1[o][0] = zx; sz1[o][1] = zy; sz1[o][2] = zz;
  }
  __syncthreads();
  float a0 = 0;
  for (int i = 0; i < 16; ++i) a0 += sz0[i] * A.wo0[i * 32 + o];
  for (int i = 0; i < A.c0i; ++i) a0 += A.f0[n * A.c0i + i] * A.wo0[(16 + i) * 32 + o];
  float a1[3] = {0, 0, 0};
  for (int i = 0; i < 16; ++i) {
    float w = A.wo1[i * 32 + o];
    a1[0] += sz1[i][0] * w; a1[1] += sz1[i][1] * w; a1[2] += sz1[i][2] * w;
  }
  if (A.has_f1) {
    for (int i = 0; i < 32; ++i) {
      float w = A.wo1[(16 + i) * 32 + o];
      const float* fp = A.f1 + (n * 32 + i) * 3;
      a1[0] += fp[0] * w; a1[1] += fp[1] * w; a1[2] += fp[2] * w;
    }
  }
  float n0 = fabsf(a0);
  float mu = wsum32(n0) * (1.0f / 32.0f);
  float dv = n0 - mu;
  float var = wsum32(dv * dv) * (1.0f / 32.0f);
  float ln = fmaxf(dv * rsqrtf(var + 1e-5f) * A.g0[o] + A.b0[o], 0.0f);
  A.f0o[n * 32 + o] = a0 * (ln / (n0 + 1e-8f));
  float n1 = sqrtf(a1[0] * a1[0] + a1[1] * a1[1] + a1[2] * a1[2] + 1e-12f);
  mu = wsum32(n1) * (1.0f / 32.0f);
  dv = n1 - mu;
  var = wsum32(dv * dv) * (1.0f / 32.0f);
  float ln1 = fmaxf(dv * rsqrtf(var + 1e-5f) * A.g1[o] + A.b1[o], 0.0f) / n1;
  for (int c = 0; c < 3; ++c) A.f1o[(n * 32 + o) * 3 + c] = a1[c] * ln1;
}

// final conv aggregation: segment-sum over incoming edges + self-interaction
__global__ __launch_bounds__(64) void final_agg_kernel(const float* m0, const float* f0,
                                                       const float* wself, float* out0) {
  int n = blockIdx.x, b = n >> 6, d = n & 63;
  int o = threadIdx.x;
  float acc = 0.0f;
  for (int sp = 0; sp < 63; ++sp) {
    int s = sp + (sp >= d ? 1 : 0);
    int dd = d - (d > s ? 1 : 0);
    int e = b * EPB + s * 63 + dd;
    acc += m0[e * 64 + o];
  }
  for (int i = 0; i < 32; ++i) acc += f0[n * 32 + i] * wself[i * 64 + o];
  out0[n * 64 + o] = acc;
}

// mean pooling over atoms + final projection
__global__ __launch_bounds__(128) void pool_proj_kernel(const float* out0, const float* pw,
                                                        const float* pb, float* out) {
  __shared__ float pooled[64];
  int b = blockIdx.x, t = threadIdx.x;
  if (t < 64) {
    float s = 0.0f;
    for (int a = 0; a < 64; ++a) s += out0[(b * 64 + a) * 64 + t];
    pooled[t] = s * (1.0f / 64.0f);
  }
  __syncthreads();
  float acc = pb[t];
  for (int o = 0; o < 64; ++o) acc += pooled[o] * pw[o * 128 + t];
  out[b * 128 + t] = acc;
}

// ---------------- host ----------------

struct RadP { const float *b1, *b2, *be, *g, *w1, *w2; };

extern "C" void kernel_launch(void* const* d_in, const int* in_sizes, int n_in,
                              void* d_out, int out_size, void* d_ws, size_t ws_size,
                              hipStream_t stream) {
  (void)in_sizes; (void)n_in; (void)out_size; (void)ws_size;
  auto rad = [&](int base) {
    RadP r;
    r.b1 = (const float*)d_in[base + 0];
    r.b2 = (const float*)d_in[base + 1];
    r.be = (const float*)d_in[base + 2];
    r.g  = (const float*)d_in[base + 3];
    r.w1 = (const float*)d_in[base + 4];
    r.w2 = (const float*)d_in[base + 5];
    return r;
  };
  const float* h = (const float*)d_in[0];
  const float* x = (const float*)d_in[1];

  char* wsc = (char*)d_ws;
  size_t off = 0;
  auto alloc = [&](size_t bytes) -> void* {
    void* p = wsc + off;
    off = (off + bytes + 255) & ~(size_t)255;
    return p;
  };
  float* rbuf  = (float*)alloc((size_t)NE * 4);
  float* dirv  = (float*)alloc((size_t)NE * 3 * 4);
  __bf16* hhS[10]; for (int i = 0; i < 10; ++i) hhS[i] = (__bf16*)alloc((size_t)NE * 32 * 2);
  __bf16* bS[10];  for (int i = 0; i < 10; ++i) bS[i]  = (__bf16*)alloc((size_t)4 * 33 * 512 * 2);
  __bf16 *Ff0 = (__bf16*)alloc((size_t)NE * 32 * 2), *Fpr = (__bf16*)alloc((size_t)NE * 32 * 2);
  __bf16 *F1x = (__bf16*)alloc((size_t)NE * 32 * 2), *F1y = (__bf16*)alloc((size_t)NE * 32 * 2);
  __bf16 *F1z = (__bf16*)alloc((size_t)NE * 32 * 2);
  __bf16 *Fcx = (__bf16*)alloc((size_t)NE * 32 * 2), *Fcy = (__bf16*)alloc((size_t)NE * 32 * 2);
  __bf16 *Fcz = (__bf16*)alloc((size_t)NE * 32 * 2);
  float* k0b = (float*)alloc((size_t)NE * 32 * 4);
  float* k1b = (float*)alloc((size_t)NE * 96 * 4);   // reused as final m0 [E][64]
  float* v0b = (float*)alloc((size_t)NE * 16 * 4);
  float* v1b = (float*)alloc((size_t)NE * 48 * 4);
  float* lgb = (float*)alloc((size_t)NE * 8 * 4);
  float* atb = (float*)alloc((size_t)NE * 8 * 4);
  float* q0b = (float*)alloc((size_t)NN * 32 * 4);
  float* q1b = (float*)alloc((size_t)NN * 96 * 4);
  float* f0p[2] = {(float*)alloc((size_t)NN * 32 * 4), (float*)alloc((size_t)NN * 32 * 4)};
  float* f1p[2] = {(float*)alloc((size_t)NN * 96 * 4), (float*)alloc((size_t)NN * 96 * 4)};
  float* out0f = (float*)alloc((size_t)NN * 64 * 4);

  geom_kernel<<<NE / 256, 256, 0, stream>>>(x, rbuf, dirv);

  const float* f0cur = h;     // layer0 input: [NN,16]
  const float* f1cur = nullptr;
  const int lbase[4] = {15, 46, 114, 182};
  dim3 hhg(NE * 32 / 256);

  for (int l = 0; l < 4; ++l) {
    const bool hf1 = (l > 0);
    const int c0i = hf1 ? 32 : 16;
    const int kb = lbase[l];
    RadP kr[5], vr[5];
    const float *wo0, *wo1, *wq0, *wq1 = nullptr, *ng0, *nb0, *ng1, *nb1;
    int nk;
    if (!hf1) {
      nk = 2;
      kr[0] = rad(kb + 0); kr[1] = rad(kb + 6);
      nb0 = (const float*)d_in[kb + 12]; nb1 = (const float*)d_in[kb + 13];
      ng0 = (const float*)d_in[kb + 14]; ng1 = (const float*)d_in[kb + 15];
      vr[0] = rad(kb + 16); vr[1] = rad(kb + 22);
      wo0 = (const float*)d_in[kb + 28]; wo1 = (const float*)d_in[kb + 29];
      wq0 = (const float*)d_in[kb + 30];
    } else {
      nk = 5;
      for (int i = 0; i < 5; ++i) kr[i] = rad(kb + i * 6);
      nb0 = (const float*)d_in[kb + 30]; nb1 = (const float*)d_in[kb + 31];
      ng0 = (const float*)d_in[kb + 32]; ng1 = (const float*)d_in[kb + 33];
      for (int i = 0; i < 5; ++i) vr[i] = rad(kb + 34 + i * 6);
      wo0 = (const float*)d_in[kb + 64]; wo1 = (const float*)d_in[kb + 65];
      wq0 = (const float*)d_in[kb + 66]; wq1 = (const float*)d_in[kb + 67];
    }

    edge_feat_kernel<<<NE / 256, 256, 0, stream>>>(f0cur, f1cur, dirv, Ff0, Fpr,
                                                   F1x, F1y, F1z, Fcx, Fcy, Fcz, c0i, hf1);
    for (int i = 0; i < nk; ++i) {
      hh_kernel<<<hhg, 256, 0, stream>>>(rbuf, kr[i].w1, kr[i].b1, kr[i].g, kr[i].be, hhS[i]);
      hh_kernel<<<hhg, 256, 0, stream>>>(rbuf, vr[i].w1, vr[i].b1, vr[i].g, vr[i].be, hhS[5 + i]);
      int cin = (hf1 || i < 2) ? c0i : 32;
      pack_b_kernel<<<33 * 2, 32, 0, stream>>>(kr[i].w2, kr[i].b2, bS[i], cin, 32, 2);
      pack_b_kernel<<<33 * 1, 32, 0, stream>>>(vr[i].w2, vr[i].b2, bS[5 + i], cin, 16, 1);
    }

    ConvArgs ck = {hhS[0], hhS[1], hhS[2], hhS[3], hhS[4],
                   bS[0], bS[1], bS[2], bS[3], bS[4],
                   Ff0, Fpr, F1x, F1y, F1z, Fcx, Fcy, Fcz,
                   dirv, k0b, k1b, 32, 2, hf1 ? 1 : 0, 0};
    conv_kernel<<<(NTILE / 4) * 2, 128, 0, stream>>>(ck);
    ConvArgs cv = {hhS[5], hhS[6], hhS[7], hhS[8], hhS[9],
                   bS[5], bS[6], bS[7], bS[8], bS[9],
                   Ff0, Fpr, F1x, F1y, F1z, Fcx, Fcy, Fcz,
                   dirv, v0b, v1b, 16, 1, hf1 ? 1 : 0, 0};
    conv_kernel<<<(NTILE / 4) * 1, 128, 0, stream>>>(cv);

    q_kernel<<<NN * 32 / 256, 256, 0, stream>>>(f0cur, f1cur, wq0, wq1, q0b, q1b, c0i, hf1);
    logits_kernel<<<NE / 256, 256, 0, stream>>>(q0b, q1b, k0b, k1b, lgb, hf1, hf1 ? 0.25f : 0.5f);
    softmax_kernel<<<NN, 64, 0, stream>>>(lgb, atb);

    AggArgs ag = {atb, v0b, v1b, f0cur, f1cur, wo0, wo1, ng0, nb0, ng1, nb1,
                  f0p[l & 1], f1p[l & 1], c0i, hf1 ? 1 : 0};
    aggregate_kernel<<<NN, 32, 0, stream>>>(ag);
    f0cur = f0p[l & 1];
    f1cur = f1p[l & 1];
  }

  // final conv (degree-0 only, cout=64) + aggregation + pooling + projection
  RadP fr00 = rad(2), fr10 = rad(8);
  const float* wself = (const float*)d_in[14];
  const float* projb = (const float*)d_in[250];
  const float* projw = (const float*)d_in[251];

  edge_feat_kernel<<<NE / 256, 256, 0, stream>>>(f0cur, f1cur, dirv, Ff0, Fpr,
                                                 F1x, F1y, F1z, Fcx, Fcy, Fcz, 32, 1);
  hh_kernel<<<hhg, 256, 0, stream>>>(rbuf, fr00.w1, fr00.b1, fr00.g, fr00.be, hhS[0]);
  hh_kernel<<<hhg, 256, 0, stream>>>(rbuf, fr10.w1, fr10.b1, fr10.g, fr10.be, hhS[1]);
  pack_b_kernel<<<33 * 4, 32, 0, stream>>>(fr00.w2, fr00.b2, bS[0], 32, 64, 4);
  pack_b_kernel<<<33 * 4, 32, 0, stream>>>(fr10.w2, fr10.b2, bS[1], 32, 64, 4);
  float* m0fin = k1b;  // [E][64] reuse
  ConvArgs cf = {hhS[0], nullptr, hhS[1], nullptr, nullptr,
                 bS[0], nullptr, bS[1], nullptr, nullptr,
                 Ff0, Fpr, F1x, F1y, F1z, Fcx, Fcy, Fcz,
                 dirv, m0fin, nullptr, 64, 4, 1, 1};
  conv_kernel<<<(NTILE / 4) * 4, 128, 0, stream>>>(cf);
  final_agg_kernel<<<NN, 64, 0, stream>>>(m0fin, f0cur, wself, out0f);
  pool_proj_kernel<<<NB, 128, 0, stream>>>(out0f, projw, projb, (float*)d_out);
}
